// ActorNetwork_68908455297461
// MI455X (gfx1250) — matrix operand order repaired and checked by reading, not compiled, not run on hardware
//
#include <hip/hip_runtime.h>
#include <math.h>

typedef __attribute__((ext_vector_type(16))) _Float16 v16h;
typedef __attribute__((ext_vector_type(8)))  float    v8f;

#define NSHEET  16
#define MAXW    128
#define MAXH    128
#define MAXP    25
#define HIDDEN  1024
#define CNN_DIM 65536
#define IN_DIM  65611            // CNN_DIM + 3*MAXP
#define KPAD    65632            // 2051 * 32
#define NKT     2051             // number of K tiles (32 wide)
#define KCH     32               // K-chunk blocks for fc1
#define TILES_PER_CH 65          // ceil(2051/32)
#define ACT_DIM 800

// workspace layout (in float/int words)
#define OFF_CONV1 0              // float: 16*2*64*64   = 131072
#define OFF_COMB  131072         // float: KPAD        = 65632
#define OFF_PART  196704         // float: 32*1024     = 32768
#define OFF_H     229472         // float: 1024
#define OFF_LOG   230496         // float: 800
#define OFF_SAT   231296         // int:   16*129*129  = 266256
#define OFF_ROW   497552         // int:   16*128*128  = 262144
#define OFF_FITS  759696         // int:   800

// ---------------- conv1: [16,1,128,128] -> relu -> [16,2,64,64] ----------------
__global__ __launch_bounds__(256) void k_conv1(const int* __restrict__ stocks,
                                               const float* __restrict__ w,
                                               const float* __restrict__ b,
                                               float* __restrict__ out) {
  int idx = blockIdx.x * 256 + threadIdx.x;      // 131072 total
  int j = idx & 63, i = (idx >> 6) & 63, o = (idx >> 12) & 1, s = idx >> 13;
  float acc = b[o];
  #pragma unroll
  for (int di = 0; di < 3; ++di)
    #pragma unroll
    for (int dj = 0; dj < 3; ++dj) {
      int ii = 2 * i - 1 + di, jj = 2 * j - 1 + dj;
      if (ii >= 0 && ii < 128 && jj >= 0 && jj < 128) {
        float x = (stocks[(s * 128 + ii) * 128 + jj] == -1) ? 1.f : 0.f;
        acc += w[o * 9 + di * 3 + dj] * x;
      }
    }
  out[idx] = fmaxf(acc, 0.f);
}

// ---------------- conv2: [16,2,64,64] -> relu -> [16,4,32,32] (into combined) ----------------
__global__ __launch_bounds__(256) void k_conv2(const float* __restrict__ in,
                                               const float* __restrict__ w,
                                               const float* __restrict__ b,
                                               float* __restrict__ comb) {
  int idx = blockIdx.x * 256 + threadIdx.x;      // 65536 total
  int j = idx & 31, i = (idx >> 5) & 31, o = (idx >> 10) & 3, s = idx >> 12;
  float acc = b[o];
  #pragma unroll
  for (int c = 0; c < 2; ++c)
    #pragma unroll
    for (int di = 0; di < 3; ++di)
      #pragma unroll
      for (int dj = 0; dj < 3; ++dj) {
        int ii = 2 * i - 1 + di, jj = 2 * j - 1 + dj;
        if (ii >= 0 && ii < 64 && jj >= 0 && jj < 64) {
          acc += w[((o * 2 + c) * 3 + di) * 3 + dj] *
                 in[((s * 2 + c) * 64 + ii) * 64 + jj];
        }
      }
  comb[idx] = fmaxf(acc, 0.f);
}

// ---------------- product-feature tail + zero pad ----------------
__global__ void k_pf(const int* __restrict__ sizes, const int* __restrict__ qty,
                     float* __restrict__ comb) {
  int t = threadIdx.x;  // 96 threads
  if (t < 3 * MAXP) {
    int p = t / 3, r = t - 3 * p;
    int v = (r == 0) ? sizes[2 * p] : (r == 1) ? sizes[2 * p + 1] : qty[p];
    comb[CNN_DIM + t] = (float)v;
  } else if (t < KPAD - CNN_DIM) {
    comb[CNN_DIM + t] = 0.f;   // pad K to multiple of 32
  }
}

// Build one 32-K WMMA step: A = x-chunk replicated into every row (all lanes
// load the same K-mapped values -> every row of D holds the same GEMV slice),
// B = fp32 weight tile converted to f16 in-register.
// A (16-bit 16x32) lane-half K map: K = t + 16*hi
// B (16-bit 32x16) lane-half K map: lanes 0-15 K=0..15, lanes 16-31 K=16..31
__device__ __forceinline__ v8f wmma_step(const float* __restrict__ x,
                                         const float* __restrict__ w,
                                         int k0, int hi, int col, int ldw,
                                         v8f c) {
  v16h a, bm;
  #pragma unroll
  for (int t = 0; t < 16; ++t) {
    int ka = k0 + t + 16 * hi;
    a[t] = (_Float16)x[ka];
    int kb = k0 + t + 16 * hi;
    bm[t] = (_Float16)w[(long)kb * ldw + col];
  }
  return __builtin_amdgcn_wmma_f32_16x16x32_f16(false, a, false, bm,
                                                (short)0, c, false, false);
}

// ---------------- fc1 via WMMA: partials[kc][1024] ----------------
// grid (4 col-groups, 32 k-chunks), 512 threads = 16 waves, wave handles 16 cols.
__global__ __launch_bounds__(512) void k_fc1(const float* __restrict__ x,
                                             const float* __restrict__ w,
                                             float* __restrict__ part) {
  int lane = threadIdx.x & 31;
  int wave = threadIdx.x >> 5;            // 0..15
  int n    = lane & 15;
  int hi   = lane >> 4;
  int col  = blockIdx.x * 256 + wave * 16 + n;
  int kt0  = blockIdx.y * TILES_PER_CH;
  int kt1  = kt0 + TILES_PER_CH; if (kt1 > NKT) kt1 = NKT;
  int ktm  = (kt1 < NKT - 1) ? kt1 : (NKT - 1);   // unguarded main range end

  v8f c = {};
  #pragma unroll 2
  for (int kt = kt0; kt < ktm; ++kt) {
    int k0 = kt * 32;
    __builtin_prefetch(&w[(long)(k0 + 32) * HIDDEN + col], 0, 0);
    c = wmma_step(x, w, k0, hi, col, HIDDEN, c);
  }
  if (kt1 == NKT) {
    // final tile (kt = NKT-1): B rows beyond IN_DIM must read as zero
    int k0 = (NKT - 1) * 32;
    v16h a, bm;
    #pragma unroll
    for (int t = 0; t < 16; ++t) {
      int ka = k0 + t + 16 * hi;
      a[t] = (_Float16)x[ka];                       // x zero-padded to KPAD
      int kb = k0 + t + 16 * hi;
      int kc = (kb < IN_DIM) ? kb : (IN_DIM - 1);
      float bv = w[(long)kc * HIDDEN + col];
      bm[t] = (_Float16)((kb < IN_DIM) ? bv : 0.f);
    }
    c = __builtin_amdgcn_wmma_f32_16x16x32_f16(false, a, false, bm,
                                               (short)0, c, false, false);
  }
  // D row M=0 lives in c[0] of lanes 0..15 (hi==0)
  if (hi == 0) part[blockIdx.y * HIDDEN + col] = c[0];
}

// ---------------- fc1 reduce + bias + relu ----------------
__global__ __launch_bounds__(256) void k_fc1red(const float* __restrict__ part,
                                                const float* __restrict__ bias,
                                                float* __restrict__ h) {
  int j = blockIdx.x * 256 + threadIdx.x;     // 1024
  float acc = bias[j];
  for (int kc = 0; kc < KCH; ++kc) acc += part[kc * HIDDEN + j];
  h[j] = fmaxf(acc, 0.f);
}

// ---------------- fc2 via WMMA: logits[800] ----------------
// 50 blocks x 32 threads (one wave -> 16 columns)
__global__ __launch_bounds__(32) void k_fc2(const float* __restrict__ h,
                                            const float* __restrict__ w,
                                            const float* __restrict__ bias,
                                            float* __restrict__ logits) {
  int lane = threadIdx.x;
  int n = lane & 15, hi = lane >> 4;
  int col = blockIdx.x * 16 + n;
  v8f c = {};
  #pragma unroll 2
  for (int kt = 0; kt < HIDDEN / 32; ++kt) {
    c = wmma_step(h, w, kt * 32, hi, col, ACT_DIM, c);
  }
  if (hi == 0) logits[col] = c[0] + bias[col];
}

// ---------------- summed-area table: row scan ----------------
__global__ __launch_bounds__(128) void k_rowscan(const int* __restrict__ stocks,
                                                 int* __restrict__ rowc) {
  int s = blockIdx.x, i = threadIdx.x;
  int base = (s * 128 + i) * 128;
  int c = 0;
  for (int j = 0; j < 128; ++j) {
    c += (stocks[base + j] != -1) ? 1 : 0;
    rowc[base + j] = c;
  }
}

// ---------------- summed-area table: column scan + zero borders ----------------
__global__ __launch_bounds__(160) void k_colscan(const int* __restrict__ rowc,
                                                 int* __restrict__ sat) {
  int s = blockIdx.x, j = threadIdx.x;
  int* S = sat + s * 129 * 129;
  if (j < 128) {
    S[j + 1] = 0;                                  // top border row 0
    int c = 0;
    for (int i = 0; i < 128; ++i) {
      c += rowc[(s * 128 + i) * 128 + j];
      S[(i + 1) * 129 + (j + 1)] = c;
    }
  } else if (j == 128) {
    for (int i = 0; i < 129; ++i) S[i * 129] = 0;  // left border col 0
  }
}

// ---------------- valid-action scan: one block per (s,p,orientation) ----------------
__global__ __launch_bounds__(256) void k_valid(const int* __restrict__ sizes,
                                               const int* __restrict__ sat,
                                               int* __restrict__ fits) {
  int id = blockIdx.x;                 // (s*25 + p)*2 + m
  int m = id & 1, sp = id >> 1, p = sp % MAXP, s = sp / MAXP;
  int w = sizes[2 * p + (m ? 1 : 0)];
  int h = sizes[2 * p + (m ? 0 : 1)];
  const int* S = sat + s * 129 * 129;
  __shared__ int flag;
  if (threadIdx.x == 0) flag = 0;
  __syncthreads();
  bool any = false;
  for (int pos = threadIdx.x; pos < 128 * 128; pos += 256) {
    int i = pos >> 7, j = pos & 127;
    if (i <= 128 - w && j <= 128 - h) {
      int i2 = i + w, j2 = j + h;
      int win = S[i2 * 129 + j2] - S[i * 129 + j2] - S[i2 * 129 + j] + S[i * 129 + j];
      if (win == 0) any = true;
    }
  }
  if (any) atomicOr(&flag, 1);
  __syncthreads();
  if (threadIdx.x == 0) fits[id] = flag;
}

// ---------------- mask + softmax ----------------
__global__ __launch_bounds__(1024) void k_softmax(const float* __restrict__ logits,
                                                  const int* __restrict__ fits,
                                                  const int* __restrict__ qty,
                                                  float* __restrict__ out) {
  __shared__ float sb[1024];
  int tid = threadIdx.x;
  float l = -3.0e38f;
  if (tid < ACT_DIM) {
    int m = tid & 1, sp = tid >> 1, p = sp % MAXP;
    int f0 = fits[sp * 2 + 0], f1 = fits[sp * 2 + 1];
    bool have = qty[p] > 0;
    bool v = (m == 0) ? (f0 && have) : (f1 && !f0 && have);
    l = v ? logits[tid] : -1.0e9f;
  }
  sb[tid] = l;
  __syncthreads();
  for (int s2 = 512; s2 > 0; s2 >>= 1) {
    if (tid < s2) sb[tid] = fmaxf(sb[tid], sb[tid + s2]);
    __syncthreads();
  }
  float mx = sb[0];
  __syncthreads();
  float e = (tid < ACT_DIM) ? __expf(l - mx) : 0.f;
  sb[tid] = e;
  __syncthreads();
  for (int s2 = 512; s2 > 0; s2 >>= 1) {
    if (tid < s2) sb[tid] += sb[tid + s2];
    __syncthreads();
  }
  float inv = 1.f / sb[0];
  if (tid < ACT_DIM) out[tid] = e * inv;
}

extern "C" void kernel_launch(void* const* d_in, const int* in_sizes, int n_in,
                              void* d_out, int out_size, void* d_ws, size_t ws_size,
                              hipStream_t stream) {
  const int*   stocks  = (const int*)d_in[0];
  const int*   sizes   = (const int*)d_in[1];
  const int*   qty     = (const int*)d_in[2];
  const float* c1w     = (const float*)d_in[3];
  const float* c1b     = (const float*)d_in[4];
  const float* c2w     = (const float*)d_in[5];
  const float* c2b     = (const float*)d_in[6];
  const float* f1w     = (const float*)d_in[7];
  const float* f1b     = (const float*)d_in[8];
  const float* f2w     = (const float*)d_in[9];
  const float* f2b     = (const float*)d_in[10];
  float* out = (float*)d_out;

  float* ws    = (float*)d_ws;
  float* conv1 = ws + OFF_CONV1;
  float* comb  = ws + OFF_COMB;
  float* part  = ws + OFF_PART;
  float* h     = ws + OFF_H;
  float* logit = ws + OFF_LOG;
  int*   sat   = (int*)d_ws + OFF_SAT;
  int*   rowc  = (int*)d_ws + OFF_ROW;
  int*   fits  = (int*)d_ws + OFF_FITS;

  // CNN front-end
  k_conv1<<<512, 256, 0, stream>>>(stocks, c1w, c1b, conv1);
  k_conv2<<<256, 256, 0, stream>>>(conv1, c2w, c2b, comb);
  k_pf<<<1, 96, 0, stream>>>(sizes, qty, comb);

  // fc1: WMMA partials over (4 col-groups x 32 k-chunks), then deterministic reduce
  k_fc1<<<dim3(4, KCH), 512, 0, stream>>>(comb, f1w, part);
  k_fc1red<<<4, 256, 0, stream>>>(part, f1b, h);

  // fc2: WMMA, 50 column tiles
  k_fc2<<<50, 32, 0, stream>>>(h, f2w, f2b, logit);

  // valid-action mask via summed-area table
  k_rowscan<<<NSHEET, 128, 0, stream>>>(stocks, rowc);
  k_colscan<<<NSHEET, 160, 0, stream>>>(rowc, sat);
  k_valid<<<ACT_DIM, 256, 0, stream>>>(sizes, sat, fits);

  // mask + softmax -> d_out
  k_softmax<<<1, 1024, 0, stream>>>(logit, fits, qty, out);
}
